// AttentionBasedRewiring_57904749084731
// MI455X (gfx1250) — matrix operand order repaired and checked by reading, not compiled, not run on hardware
//
#include <hip/hip_runtime.h>
#include <hip/hip_bf16.h>

typedef __bf16 bf16_t;
typedef __attribute__((ext_vector_type(16))) __bf16 v16bf;
typedef __attribute__((ext_vector_type(8)))  __bf16 v8bf;
typedef __attribute__((ext_vector_type(8)))  float  v8f;
typedef __attribute__((ext_vector_type(4)))  unsigned int v4u;
typedef __attribute__((ext_vector_type(8)))  int v8i;
typedef __attribute__((ext_vector_type(4)))  int v4i;

#define NNODE 65536
#define DIN   512
#define HD    128
#define TOPK  8
#define CT    128              // column (node) tile staged in LDS per block
#define NCT   (NNODE / CT)     // 512
#define LDK_STRIDE 136         // padded row stride (elements): 64 DW data + 4 DW pad

union AFrag { v16bf v; v8bf h[2]; };

static __device__ __forceinline__ v8f wmma_bf16(v16bf a, v16bf b, v8f c) {
  // D = A(16x32 bf16) * B(32x16 bf16) + C(16x16 f32)
  return __builtin_amdgcn_wmma_f32_16x16x32_bf16(false, a, false, b, (short)0, c,
                                                 false, false);
}

// ---------------------------------------------------------------------------
// TDM: async DMA of a 128x128 bf16 tile (row stride HD) into LDS with padding
// pad_interval = 5 -> pad after every 64 DWORDs (= one 128-elem bf16 row)
// pad_amount   = 3 -> 4 DWORDs pad  => LDS row stride 68 DW = 136 elements
// ---------------------------------------------------------------------------
static __device__ __forceinline__ void tdm_load_tile(const bf16_t* gsrc,
                                                     unsigned lds_byte_addr) {
  const unsigned long long ga = (unsigned long long)(size_t)gsrc;
  v4u g0;
  g0[0] = 1u;                                   // count=1, user mode, no gather
  g0[1] = lds_byte_addr;                        // LDS destination (bytes)
  g0[2] = (unsigned)(ga & 0xFFFFFFFFu);         // global_addr[31:0]
  g0[3] = (unsigned)((ga >> 32) & 0x01FFFFFFu)  // global_addr[56:32]
        | (2u << 30);                           // type = 2 ("image")
  v8i g1;
  g1[0] = (int)((1u << 16)      // data_size = 2 bytes
              | (1u << 20)      // pad_enable
              | (5u << 22)      // pad_interval: 2^(5+1) = 64 DWORDs
              | (3u << 25));    // pad_amount: 3+1 = 4 DWORDs
  g1[1] = (int)((unsigned)HD << 16);            // tensor_dim0[15:0] = 128
  g1[2] = 0;                                    // tensor_dim0[31:16]=0, tensor_dim1[15:0]=0
  g1[3] = (int)(1u | ((unsigned)CT << 16));     // tensor_dim1[31:16]=1 (=65536), tile_dim0=128
  g1[4] = CT;                                   // tile_dim1 = 128, tile_dim2 = 0
  g1[5] = HD;                                   // tensor_dim0_stride[31:0] = 128
  g1[6] = 0;
  g1[7] = 0;
  const v4i z4 = {0, 0, 0, 0};
#if __clang_major__ >= 23
  const v8i z8 = {0, 0, 0, 0, 0, 0, 0, 0};
  __builtin_amdgcn_tensor_load_to_lds(g0, g1, z4, z4, z8, 0);
#else
  __builtin_amdgcn_tensor_load_to_lds(g0, g1, z4, z4, 0);
#endif
}

// ---------------------------------------------------------------------------
// Kernel 1: transpose + convert W [DIN, HD] f32  ->  Wt [HD, DIN] bf16
// ---------------------------------------------------------------------------
__global__ void wtrans_kernel(const float* __restrict__ W, bf16_t* __restrict__ Wt) {
  const int i = blockIdx.x * blockDim.x + threadIdx.x;   // 0 .. DIN*HD-1
  const int f = i >> 7;        // feature (row of W)
  const int h = i & (HD - 1);  // output channel
  Wt[(size_t)h * DIN + f] = (bf16_t)W[i];
}

// ---------------------------------------------------------------------------
// Kernel 2: q = x@Wq + bq, k = x@Wk + bk  (WMMA bf16, f32 accumulate)
// ---------------------------------------------------------------------------
__global__ __launch_bounds__(256) void project_kernel(
    const float* __restrict__ x,
    const bf16_t* __restrict__ Wqt, const float* __restrict__ bq,
    const bf16_t* __restrict__ Wkt, const float* __restrict__ bk,
    bf16_t* __restrict__ qbf, bf16_t* __restrict__ kbf) {
  const int wave = threadIdx.x >> 5;
  const int lane = threadIdx.x & 31;
  const int lrow = lane & 15;
  const int hi   = lane >> 4;
  const int rowbase = blockIdx.x * 128 + wave * 16;
  const float* xrow = x + (size_t)(rowbase + lrow) * DIN;

  for (int ht = 0; ht < 8; ++ht) {
    v8f cq = {};
    v8f ck = {};
    const int hcol = ht * 16 + lrow;
    const bf16_t* wq = Wqt + (size_t)hcol * DIN;
    const bf16_t* wk = Wkt + (size_t)hcol * DIN;
#pragma unroll 4
    for (int kk = 0; kk < 16; ++kk) {
      AFrag a;
      const v8f x0 = *(const v8f*)(xrow + kk * 32 + hi * 16);
      const v8f x1 = *(const v8f*)(xrow + kk * 32 + hi * 16 + 8);
#pragma unroll
      for (int e = 0; e < 8; ++e) {
        a.h[0][e] = (bf16_t)x0[e];
        a.h[1][e] = (bf16_t)x1[e];
      }
      const v16bf bqv = *(const v16bf*)(wq + kk * 32 + hi * 16);
      const v16bf bkv = *(const v16bf*)(wk + kk * 32 + hi * 16);
      cq = wmma_bf16(a.v, bqv, cq);
      ck = wmma_bf16(a.v, bkv, ck);
    }
    const float bqs = bq[hcol];
    const float bks = bk[hcol];
#pragma unroll
    for (int v = 0; v < 8; ++v) {
      const int row = rowbase + v + 8 * hi;
      qbf[(size_t)row * HD + hcol] = (bf16_t)(cq[v] + bqs);
      kbf[(size_t)row * HD + hcol] = (bf16_t)(ck[v] + bks);
    }
  }
}

// ---------------------------------------------------------------------------
// Kernel 3: fused sim = q @ k^T and per-row top-8, TDM double-buffered k-tiles
// ---------------------------------------------------------------------------
__global__ __launch_bounds__(256) void simtopk_kernel(
    const bf16_t* __restrict__ qbf, const bf16_t* __restrict__ kbf,
    float* __restrict__ ovals, int* __restrict__ oidx) {
  __shared__ __attribute__((aligned(32))) bf16_t ldsK[2][CT * LDK_STRIDE];  // 2x34 KB
  __shared__ float ldsC[8][16][17];    // per-wave C transpose scratch (padded)
  __shared__ float mrgV[8][16][TOPK];  // half-list merge scratch
  __shared__ int   mrgI[8][16][TOPK];

  const int wave = threadIdx.x >> 5;
  const int lane = threadIdx.x & 31;
  const int lrow = lane & 15;
  const int hi   = lane >> 4;
  const int rowbase = blockIdx.x * 128 + wave * 16;

  const unsigned ldsK_addr[2] = {
      (unsigned)(size_t)(&ldsK[0][0]),
      (unsigned)(size_t)(&ldsK[1][0]),
  };

  // Load this wave's A fragments (16 q-rows, K=128) once; reused for all tiles.
  AFrag a[4];
  const bf16_t* qrow = qbf + (size_t)(rowbase + lrow) * HD;
#pragma unroll
  for (int kk = 0; kk < 4; ++kk) {
    a[kk].h[0] = *(const v8bf*)(qrow + kk * 32 + hi * 16);
    a[kk].h[1] = *(const v8bf*)(qrow + kk * 32 + hi * 16 + 8);
  }

  float tv[TOPK];
  int   ti[TOPK];
#pragma unroll
  for (int j = 0; j < TOPK; ++j) { tv[j] = -3.402823466e38f; ti[j] = 0; }

  // Prologue: wave 0 kicks off the TDM DMA for tile 0 into buffer 0.
  if (wave == 0) tdm_load_tile(kbf, ldsK_addr[0]);

  for (int ct = 0; ct < NCT; ++ct) {
    const int cur = ct & 1;
    if (wave == 0) __builtin_amdgcn_s_wait_tensorcnt(0);  // tile ct landed
    __syncthreads();  // publish tile ct; all waves done reading buffer 1-cur
    if (wave == 0 && (ct + 1) < NCT)
      tdm_load_tile(kbf + (size_t)(ct + 1) * CT * HD, ldsK_addr[1 - cur]);
    // Cache-prime two tiles ahead (global_prefetch_b8).
    __builtin_prefetch(kbf + (size_t)((ct + 2) & (NCT - 1)) * CT * HD +
                           threadIdx.x * 64, 0, 1);
    const bf16_t* kt = &ldsK[cur][0];

#pragma unroll 2
    for (int sub = 0; sub < 8; ++sub) {
      v8f c = {};
#pragma unroll
      for (int kk = 0; kk < 4; ++kk) {
        const bf16_t* bp = kt + (sub * 16 + lrow) * LDK_STRIDE + kk * 32 + hi * 16;
        AFrag b;
        b.h[0] = *(const v8bf*)(bp);
        b.h[1] = *(const v8bf*)(bp + 8);
        c = wmma_bf16(a[kk].v, b.v, c);
      }
      // Transpose C through LDS: lane holds column lrow, rows v+8*hi.
      __builtin_amdgcn_wave_barrier();
#pragma unroll
      for (int v = 0; v < 8; ++v) ldsC[wave][v + 8 * hi][lrow] = c[v];
      __builtin_amdgcn_wave_barrier();
      const int colbase = ct * CT + sub * 16 + hi * 8;
#pragma unroll
      for (int j = 0; j < 8; ++j) {
        const float sv = ldsC[wave][lrow][hi * 8 + j];
        if (sv > tv[TOPK - 1]) {
          tv[TOPK - 1] = sv;
          ti[TOPK - 1] = colbase + j;
#pragma unroll
          for (int s = TOPK - 1; s > 0; --s) {
            if (tv[s] > tv[s - 1]) {
              const float tf = tv[s]; tv[s] = tv[s - 1]; tv[s - 1] = tf;
              const int   tt = ti[s]; ti[s] = ti[s - 1]; ti[s - 1] = tt;
            }
          }
        }
      }
      __builtin_amdgcn_wave_barrier();
    }
  }

  // Merge the two per-row half-lists (lane lrow+16 -> lane lrow) and emit.
  if (hi == 1) {
#pragma unroll
    for (int j = 0; j < TOPK; ++j) { mrgV[wave][lrow][j] = tv[j]; mrgI[wave][lrow][j] = ti[j]; }
  }
  __syncthreads();
  if (hi == 0) {
#pragma unroll
    for (int j = 0; j < TOPK; ++j) {
      const float sv = mrgV[wave][lrow][j];
      const int   si = mrgI[wave][lrow][j];
      if (sv > tv[TOPK - 1]) {
        tv[TOPK - 1] = sv;
        ti[TOPK - 1] = si;
#pragma unroll
        for (int s = TOPK - 1; s > 0; --s) {
          if (tv[s] > tv[s - 1]) {
            const float tf = tv[s]; tv[s] = tv[s - 1]; tv[s - 1] = tf;
            const int   tt = ti[s]; ti[s] = ti[s - 1]; ti[s - 1] = tt;
          }
        }
      }
    }
    const int row = rowbase + lrow;
#pragma unroll
    for (int j = 0; j < TOPK; ++j) {
      ovals[(size_t)row * TOPK + j] = tv[j];
      oidx [(size_t)row * TOPK + j] = ti[j];
    }
  }
}

// ---------------------------------------------------------------------------
extern "C" void kernel_launch(void* const* d_in, const int* in_sizes, int n_in,
                              void* d_out, int out_size, void* d_ws, size_t ws_size,
                              hipStream_t stream) {
  (void)in_sizes; (void)n_in; (void)out_size; (void)ws_size;
  const float* x  = (const float*)d_in[0];
  const float* Wq = (const float*)d_in[1];
  const float* bq = (const float*)d_in[2];
  const float* Wk = (const float*)d_in[3];
  const float* bk = (const float*)d_in[4];

  // Workspace layout (bf16): q [N,H], k [N,H], Wq^T [H,D], Wk^T [H,D]  ~ 34 MB
  bf16_t* qbf = (bf16_t*)d_ws;
  bf16_t* kbf = qbf + (size_t)NNODE * HD;
  bf16_t* Wqt = kbf + (size_t)NNODE * HD;
  bf16_t* Wkt = Wqt + (size_t)HD * DIN;

  // Output: vals [N, 8] f32 followed by idx [N, 8] i32.
  float* ovals = (float*)d_out;
  int*   oidx  = (int*)d_out + (size_t)NNODE * TOPK;

  wtrans_kernel<<<(DIN * HD) / 256, 256, 0, stream>>>(Wq, Wqt);
  wtrans_kernel<<<(DIN * HD) / 256, 256, 0, stream>>>(Wk, Wkt);
  project_kernel<<<NNODE / 128, 256, 0, stream>>>(x, Wqt, bq, Wkt, bk, qbf, kbf);
  simtopk_kernel<<<NNODE / 128, 256, 0, stream>>>(qbf, kbf, ovals, oidx);
}
